// AdjMultiHeadAttention_64132451664328
// MI455X (gfx1250) — matrix operand order repaired and checked by reading, not compiled, not run on hardware
//
#include <hip/hip_runtime.h>
#include <hip/hip_bf16.h>
#include <stdint.h>

// ---------- types ----------
typedef __attribute__((ext_vector_type(16))) __bf16 v16bf;
typedef __attribute__((ext_vector_type(8)))  float  v8f;

struct __align__(16) u4 { unsigned int x, y, z, w; };

union Frag { v16bf v; u4 q[2]; };

__device__ __forceinline__ unsigned short f2bf(float f) {
    unsigned int u = __float_as_uint(f);
    u += 0x7fffu + ((u >> 16) & 1u);            // round-to-nearest-even
    return (unsigned short)(u >> 16);
}

// ---------- CDNA5 async global->LDS (ASYNCcnt path, bypasses VGPRs) ----------
__device__ __forceinline__ void async_g2l_b128(void* lds_ptr, const void* gptr) {
    const unsigned loff = (unsigned)(uintptr_t)lds_ptr;   // low 32b = LDS address
    asm volatile("global_load_async_to_lds_b128 %0, %1, off"
                 :: "v"(loff), "v"(gptr) : "memory");
}
__device__ __forceinline__ void wait_async0() {
    asm volatile("s_wait_asynccnt 0" ::: "memory");
}

// ---------- constants ----------
#define B_   32
#define N_   512
#define E_   768
#define H_   12
#define HD_  64
#define QKVW 2304          // 3*E, row width of qkv buffer
#define M_   (B_ * N_)     // 16384

// ---------- fp32 -> bf16 conversion ----------
__global__ void cvt_bf16_kernel(const float* __restrict__ in,
                                unsigned short* __restrict__ out, int n) {
    int i = blockIdx.x * blockDim.x + threadIdx.x;
    if (i < n) out[i] = f2bf(in[i]);
}

// ---------- bf16 WMMA GEMM: C[M,Nc] = A[M,K] * Bt[Nc,K]^T + bias ----------
// Workgroup tile 128x128, 8 waves in 4(row) x 2(col), each wave 32x64 (2x4 WMMA tiles).
// K-tiles double-buffered in LDS, staged with global_load_async_to_lds_b128.
template <int OUT_BF16>
__global__ void __launch_bounds__(256)
gemm_bf16_wmma(const unsigned short* __restrict__ A,
               const unsigned short* __restrict__ Bt,
               const float* __restrict__ bias,
               void* __restrict__ Cout,
               int M, int Ncols, int K) {
    __shared__ __align__(16) unsigned short As[2][128][40];
    __shared__ __align__(16) unsigned short Bs[2][128][40];

    const int tid = threadIdx.x;
    const int w = tid >> 5, l = tid & 31, l16 = l & 15, lh = l >> 4;
    const int wr = w & 3, wc = w >> 2;
    const int rowBase = blockIdx.y * 128;
    const int colBase = blockIdx.x * 128;

    // cooperative tile staging: each thread 16 bf16 of A and of Bt (2 async b128 each)
    const int ldr = tid >> 1;
    const int ldh = (tid & 1) * 16;
    const unsigned short* gA = A  + (size_t)(rowBase + ldr) * K + ldh;
    const unsigned short* gB = Bt + (size_t)(colBase + ldr) * K + ldh;

    v8f acc[2][4];
#pragma unroll
    for (int rt = 0; rt < 2; ++rt)
#pragma unroll
        for (int ct = 0; ct < 4; ++ct) acc[rt][ct] = (v8f){0,0,0,0,0,0,0,0};

    const int nk = K >> 5;
    // preload k-tile 0
    async_g2l_b128(&As[0][ldr][ldh],     gA);
    async_g2l_b128(&As[0][ldr][ldh + 8], gA + 8);
    async_g2l_b128(&Bs[0][ldr][ldh],     gB);
    async_g2l_b128(&Bs[0][ldr][ldh + 8], gB + 8);
    wait_async0();
    __syncthreads();

    for (int kt = 0; kt < nk; ++kt) {
        const int cur = kt & 1;
        if (kt + 1 < nk) {   // async-prefetch next k-tile into the other buffer
            const int nxt = cur ^ 1;
            const unsigned short* pA = gA + (size_t)(kt + 1) * 32;
            const unsigned short* pB = gB + (size_t)(kt + 1) * 32;
            async_g2l_b128(&As[nxt][ldr][ldh],     pA);
            async_g2l_b128(&As[nxt][ldr][ldh + 8], pA + 8);
            async_g2l_b128(&Bs[nxt][ldr][ldh],     pB);
            async_g2l_b128(&Bs[nxt][ldr][ldh + 8], pB + 8);
        }
        Frag fa[2];
#pragma unroll
        for (int rt = 0; rt < 2; ++rt) {
            const unsigned short* ap = &As[cur][wr * 32 + rt * 16 + l16][8 * lh];
            fa[rt].q[0] = *(const u4*)ap;          // k = 8*lh .. +7
            fa[rt].q[1] = *(const u4*)(ap + 16);   // k = 16+8*lh .. +7
        }
#pragma unroll
        for (int ct = 0; ct < 4; ++ct) {
            Frag fb;
            const unsigned short* bp = &Bs[cur][wc * 64 + ct * 16 + l16][8 * lh];
            fb.q[0] = *(const u4*)bp;              // k = 8*lh .. +7
            fb.q[1] = *(const u4*)(bp + 16);       // k = 16+8*lh .. +7
#pragma unroll
            for (int rt = 0; rt < 2; ++rt)
                acc[rt][ct] = __builtin_amdgcn_wmma_f32_16x16x32_bf16(
                    false, fa[rt].v, false, fb.v, (short)0, acc[rt][ct], false, false);
        }
        wait_async0();
        __syncthreads();
    }

#pragma unroll
    for (int rt = 0; rt < 2; ++rt)
#pragma unroll
        for (int ct = 0; ct < 4; ++ct)
#pragma unroll
            for (int i = 0; i < 8; ++i) {
                const int row = rowBase + wr * 32 + rt * 16 + i + 8 * lh;
                const int col = colBase + wc * 64 + ct * 16 + l16;
                const float v = acc[rt][ct][i] + bias[col];
                if (OUT_BF16)
                    ((unsigned short*)Cout)[(size_t)row * Ncols + col] = f2bf(v);
                else
                    ((float*)Cout)[(size_t)row * Ncols + col] = v;
            }
}

// ---------- fused attention ----------
// One workgroup = (b, h, 64-query-row block). 512 threads = 16 waves.
// Wave w: qt = w>>2 (16-row tile), ch = w&3 (128-key column quarter) -> 8 score tiles
// (64 accum VGPRs/wave -> no spills). Softmax: shfl reduce + 4-way LDS combine.
// PV: each wave computes exactly one 16x16 output tile from P (bf16 LDS) and
// V staged transposed in LDS.
#define P_LD  520            // 512 + 8 pad (shorts), row = 1040 B (16B aligned)
#define ATTN_SMEM ((64 * P_LD * 2) * 2 + (256 + 256 + 64 + 8) * 4)

__global__ void __launch_bounds__(512)
attn_fused_kernel(const unsigned short* __restrict__ qkv,  // [M, 2304] bf16 (q|k|v)
                  const int* __restrict__ adj,             // [B,N,N]
                  const int* __restrict__ bond,            // [B,N,N]
                  const float* __restrict__ bond_table,    // [5]
                  unsigned short* __restrict__ ao) {       // [M, 768] bf16
    extern __shared__ __align__(16) unsigned char smem[];
    unsigned short* P    = (unsigned short*)smem;      // [64][P_LD]
    unsigned short* Vt   = P + 64 * P_LD;              // [64][P_LD]  (Vt[d][key])
    float* redmax = (float*)(Vt + 64 * P_LD);          // [4][64]
    float* redsum = redmax + 256;                      // [4][64]
    float* inv    = redsum + 256;                      // [64]
    float* btab   = inv + 64;                          // [8]

    const int wg = blockIdx.x;
    const int rb = wg & 7;                 // 64-row block
    const int h  = (wg >> 3) % H_;
    const int b  = wg / (8 * H_);

    const int tid = threadIdx.x;
    const int w = tid >> 5, l = tid & 31, l16 = l & 15, lh = l >> 4;
    const int qt = w >> 2;        // 0..3: 16-row query tile
    const int ch = w & 3;         // 0..3: 128-key column quarter

    if (tid < 8) btab[tid] = (tid >= 1 && tid <= 4) ? bond_table[tid] : 0.0f;

    // ---- stage V transposed into LDS: Vt[d][key] ----
    {
        const unsigned short* Vg = qkv + (size_t)(b * N_) * QKVW + 1536 + h * HD_;
#pragma unroll
        for (int j = 0; j < 8; ++j) {
            const int it  = tid + 512 * j;       // 0..4095
            const int key = it >> 3;             // 0..511
            const int d0  = (it & 7) * 8;        // 0..56
            u4 vv = *(const u4*)(Vg + (size_t)key * QKVW + d0);
            const unsigned short* pv = (const unsigned short*)&vv;
#pragma unroll
            for (int e = 0; e < 8; ++e) Vt[(d0 + e) * P_LD + key] = pv[e];
        }
    }
    __syncthreads();

    // ---- Phase 1: scores S = Q K^T ----
    Frag fa0, fa1;
    {
        const unsigned short* Qg =
            qkv + (size_t)(b * N_ + rb * 64 + qt * 16 + l16) * QKVW + h * HD_;
        fa0.q[0] = *(const u4*)(Qg + 8 * lh);
        fa0.q[1] = *(const u4*)(Qg + 16 + 8 * lh);
        fa1.q[0] = *(const u4*)(Qg + 32 + 8 * lh);
        fa1.q[1] = *(const u4*)(Qg + 48 + 8 * lh);
    }

    v8f sacc[8];
#pragma unroll
    for (int t = 0; t < 8; ++t) sacc[t] = (v8f){0,0,0,0,0,0,0,0};

    {
        const unsigned short* Kg =
            qkv + (size_t)(b * N_ + ch * 128 + l16) * QKVW + E_ + h * HD_;
#pragma unroll
        for (int t = 0; t < 8; ++t) {
            const unsigned short* kp = Kg + (size_t)t * 16 * QKVW;
            Frag fb0, fb1;
            fb0.q[0] = *(const u4*)(kp + 8 * lh);
            fb0.q[1] = *(const u4*)(kp + 16 + 8 * lh);
            fb1.q[0] = *(const u4*)(kp + 32 + 8 * lh);
            fb1.q[1] = *(const u4*)(kp + 48 + 8 * lh);
            sacc[t] = __builtin_amdgcn_wmma_f32_16x16x32_bf16(
                false, fa0.v, false, fb0.v, (short)0, sacc[t], false, false);
            sacc[t] = __builtin_amdgcn_wmma_f32_16x16x32_bf16(
                false, fa1.v, false, fb1.v, (short)0, sacc[t], false, false);
        }
    }

    // ---- Phase 2: bias + mask + row max ----
    // element (t,i): row = rb*64+qt*16+i+8*lh, col = ch*128+t*16+l16
    const int* adjp  = adj  + ((size_t)(b * N_ + rb * 64 + qt * 16 + 8 * lh)) * N_
                            + ch * 128 + l16;
    const int* bondp = bond + ((size_t)(b * N_ + rb * 64 + qt * 16 + 8 * lh)) * N_
                            + ch * 128 + l16;
    float rowm[8];
#pragma unroll
    for (int i = 0; i < 8; ++i) rowm[i] = -3.0e38f;

#pragma unroll
    for (int t = 0; t < 8; ++t) {
#pragma unroll
        for (int i = 0; i < 8; ++i) {
            float s = sacc[t][i] * 0.125f
                    + (float)adjp[i * N_ + t * 16]
                    + btab[bondp[i * N_ + t * 16]];
            s = (s == 0.0f) ? -1.0e9f : s;
            sacc[t][i] = s;
            rowm[i] = fmaxf(rowm[i], s);
        }
    }
#pragma unroll
    for (int i = 0; i < 8; ++i) {
        float m = rowm[i];
        m = fmaxf(m, __shfl_xor(m, 1, 32));
        m = fmaxf(m, __shfl_xor(m, 2, 32));
        m = fmaxf(m, __shfl_xor(m, 4, 32));
        m = fmaxf(m, __shfl_xor(m, 8, 32));
        if (l16 == 0) redmax[ch * 64 + qt * 16 + i + 8 * lh] = m;
    }
    __syncthreads();
#pragma unroll
    for (int i = 0; i < 8; ++i) {
        const int rr = qt * 16 + i + 8 * lh;
        rowm[i] = fmaxf(fmaxf(redmax[rr], redmax[64 + rr]),
                        fmaxf(redmax[128 + rr], redmax[192 + rr]));
    }

    // ---- exp, row sums, store P (bf16) ----
    float rowsum[8];
#pragma unroll
    for (int i = 0; i < 8; ++i) rowsum[i] = 0.0f;
#pragma unroll
    for (int t = 0; t < 8; ++t) {
#pragma unroll
        for (int i = 0; i < 8; ++i) {
            const float p = __expf(sacc[t][i] - rowm[i]);
            rowsum[i] += p;
            P[(qt * 16 + i + 8 * lh) * P_LD + ch * 128 + t * 16 + l16] = f2bf(p);
        }
    }
#pragma unroll
    for (int i = 0; i < 8; ++i) {
        float s = rowsum[i];
        s += __shfl_xor(s, 1, 32);
        s += __shfl_xor(s, 2, 32);
        s += __shfl_xor(s, 4, 32);
        s += __shfl_xor(s, 8, 32);
        if (l16 == 0) redsum[ch * 64 + qt * 16 + i + 8 * lh] = s;
    }
    __syncthreads();
    if (tid < 64)
        inv[tid] = 1.0f / (redsum[tid] + redsum[64 + tid]
                         + redsum[128 + tid] + redsum[192 + tid]);
    __syncthreads();

    // ---- Phase 3: O = P @ V (one 16x16 tile per wave) ----
    const int rt = w >> 2;            // 16-row output tile
    const int ct = w & 3;             // 16-col tile of hd
    v8f acc2 = (v8f){0,0,0,0,0,0,0,0};

#pragma unroll
    for (int kk = 0; kk < 16; ++kk) {
        Frag fa, fb;
        const unsigned short* pa = &P[(rt * 16 + l16) * P_LD + 32 * kk + 8 * lh];
        fa.q[0] = *(const u4*)pa;
        fa.q[1] = *(const u4*)(pa + 16);
        const unsigned short* vp = &Vt[(ct * 16 + l16) * P_LD + 32 * kk + 8 * lh];
        fb.q[0] = *(const u4*)vp;
        fb.q[1] = *(const u4*)(vp + 16);
        acc2 = __builtin_amdgcn_wmma_f32_16x16x32_bf16(
            false, fa.v, false, fb.v, (short)0, acc2, false, false);
    }

    unsigned short* aop =
        ao + (size_t)(b * N_ + rb * 64 + rt * 16 + 8 * lh) * E_ + h * HD_ + ct * 16 + l16;
#pragma unroll
    for (int i = 0; i < 8; ++i) {
        const float v = acc2[i] * inv[rt * 16 + i + 8 * lh];
        aop[(size_t)i * E_] = f2bf(v);
    }
}

// ---------- host launcher ----------
extern "C" void kernel_launch(void* const* d_in, const int* in_sizes, int n_in,
                              void* d_out, int out_size, void* d_ws, size_t ws_size,
                              hipStream_t stream) {
    const float* x          = (const float*)d_in[0];
    const int*   adj        = (const int*)d_in[1];
    const int*   bond       = (const int*)d_in[2];
    const float* qkv_w      = (const float*)d_in[4];
    const float* qkv_b      = (const float*)d_in[5];
    const float* out_w      = (const float*)d_in[6];
    const float* out_b      = (const float*)d_in[7];
    const float* bond_table = (const float*)d_in[8];

    unsigned char* ws = (unsigned char*)d_ws;
    unsigned short* x_bf    = (unsigned short*)ws;  ws += (size_t)M_ * E_ * 2;
    unsigned short* qkvw_bf = (unsigned short*)ws;  ws += (size_t)QKVW * E_ * 2;
    unsigned short* outw_bf = (unsigned short*)ws;  ws += (size_t)E_ * E_ * 2;
    unsigned short* qkv_bf  = (unsigned short*)ws;  ws += (size_t)M_ * QKVW * 2;
    unsigned short* ao_bf   = (unsigned short*)ws;  ws += (size_t)M_ * E_ * 2;

    // 1) fp32 -> bf16 conversions
    {
        int n = M_ * E_;
        cvt_bf16_kernel<<<(n + 255) / 256, 256, 0, stream>>>(x, x_bf, n);
        n = QKVW * E_;
        cvt_bf16_kernel<<<(n + 255) / 256, 256, 0, stream>>>(qkv_w, qkvw_bf, n);
        n = E_ * E_;
        cvt_bf16_kernel<<<(n + 255) / 256, 256, 0, stream>>>(out_w, outw_bf, n);
    }

    // 2) QKV projection: qkv = x @ qkv_w^T + qkv_b  (bf16 out)
    gemm_bf16_wmma<1><<<dim3(QKVW / 128, M_ / 128), 256, 0, stream>>>(
        x_bf, qkvw_bf, qkv_b, (void*)qkv_bf, M_, QKVW, E_);

    // 3) fused attention (adj + bond bias, mask, softmax, P@V)  (bf16 out)
    attn_fused_kernel<<<dim3(B_ * H_ * (N_ / 64)), 512, ATTN_SMEM, stream>>>(
        qkv_bf, adj, bond, bond_table, ao_bf);

    // 4) output projection: out = ao @ out_w^T + out_b  (f32 out)
    gemm_bf16_wmma<0><<<dim3(E_ / 128, M_ / 128), 256, 0, stream>>>(
        ao_bf, outw_bf, out_b, d_out, M_, E_, E_);
}
